// HNN_layer_4939212391241
// MI455X (gfx1250) — matrix operand order repaired and checked by reading, not compiled, not run on hardware
//
#include <hip/hip_runtime.h>
#include <hip/hip_bf16.h>

// ---------------------------------------------------------------------------
// Fused block-diagonal 2-layer head:
//   h = relu(x @ blockdiag(kernels) + biases);  out = sigmoid(h @ fk + fb)
// BATCH = 1048576, D_IN = 128, 22 groups.
// Memory-bound (512 MB stream of x, ~22us floor at 23.3 TB/s) -> one pass of
// b128 loads, f16 WMMA (f32 accum) for the matvec, full epilogue fusion.
// ---------------------------------------------------------------------------

typedef _Float16 v16h __attribute__((ext_vector_type(16)));
typedef float    v8f  __attribute__((ext_vector_type(8)));

#define D_IN   128
#define NGROUP 22

// Prefix offsets of FEATURE_LIST = [10,13,13,7,3,6,3,13,5,4,6,4,5,4,4,5,4,3,3,7,3,3]
__constant__ int g_start_c[NGROUP] = {0,10,23,36,43,46,52,55,68,73,77,83,87,92,96,100,105,109,112,115,122,125};
__constant__ int g_end_c[NGROUP]   = {10,23,36,43,46,52,55,68,73,77,83,87,92,96,100,105,109,112,115,122,125,128};

__device__ __forceinline__ float sigmoidf(float z) {
    return 1.0f / (1.0f + __expf(-z));
}

__global__ __launch_bounds__(256) void hnn_fused_wmma(
    const float* __restrict__ x,        // [B,128] row-major
    const float* __restrict__ kernels,  // [128]
    const float* __restrict__ biases,   // [22]
    const float* __restrict__ fk,       // [22] final kernel
    const float* __restrict__ fb,       // [1]  final bias
    float*       __restrict__ out,      // [B]
    int ntiles)                         // B / 16
{
    const int lane   = threadIdx.x & 31;
    const int wave   = blockIdx.x * (blockDim.x >> 5) + (threadIdx.x >> 5);
    const int nwaves = gridDim.x * (blockDim.x >> 5);
    const int col    = lane & 15;   // WMMA B/C column owned by this lane
    const int ksel   = lane >> 4;   // which K-half of A/B this lane holds

    // ---- Build block-diagonal weight matrix in WMMA-B register layout ----
    // B tile t covers output columns t*16 .. t*16+15 (cols >= 22 are zero pad).
// Lane n holds column n; lanes 0-15 hold K = k0..k0+7, k0+16..k0+23, lanes 16-31 hold
// K = k0+8..k0+15, k0+24..k0+31 (k0 = 32*chunk).
    v16h Bm[2][4];
    #pragma unroll
    for (int t = 0; t < 2; ++t) {
        const int colg = t * 16 + col;
        int s = 1 << 30, e = 0;                       // empty range for pad cols
        if (colg < NGROUP) { s = g_start_c[colg]; e = g_end_c[colg]; }
        #pragma unroll
        for (int c = 0; c < 4; ++c) {
            #pragma unroll
            for (int i = 0; i < 16; ++i) {
                const int K = c * 32 + ksel * 8 + i + (i >= 8 ? 8 : 0);
                const float w = (K >= s && K < e) ? kernels[K] : 0.0f;
                Bm[t][c][i] = (_Float16)w;
            }
        }
    }

    // Per-lane second-layer constants (lane covers global cols `col` and `col+16`).
    const float b0  = biases[col];
    const float fk0 = fk[col];
    float b1 = 0.0f, fk1 = 0.0f;
    if (col + 16 < NGROUP) { b1 = biases[col + 16]; fk1 = fk[col + 16]; }
    const float fbias = fb[0];

    const int koff = ksel * 8;   // A-layout: lane half selects K sub-block of 8

    for (int tile = wave; tile < ntiles; tile += nwaves) {
        const float* xr = x + (size_t)(tile * 16 + col) * D_IN;

        v8f acc0 = {0.f,0.f,0.f,0.f,0.f,0.f,0.f,0.f};
        v8f acc1 = {0.f,0.f,0.f,0.f,0.f,0.f,0.f,0.f};

        #pragma unroll
        for (int c = 0; c < 4; ++c) {
            // A 16x32 f16 layout: this lane needs x[row][k0+koff .. +7] and
            // x[row][k0+koff+16 .. +23] -> two pairs of b128 loads.
            const float4* p = (const float4*)(xr + c * 32 + koff);
            const float4 lo0 = p[0];
            const float4 lo1 = p[1];
            const float4 hi0 = p[4];   // +16 floats
            const float4 hi1 = p[5];   // +20 floats

            v16h a;
            a[0]  = (_Float16)lo0.x;  a[1]  = (_Float16)lo0.y;
            a[2]  = (_Float16)lo0.z;  a[3]  = (_Float16)lo0.w;
            a[4]  = (_Float16)lo1.x;  a[5]  = (_Float16)lo1.y;
            a[6]  = (_Float16)lo1.z;  a[7]  = (_Float16)lo1.w;
            a[8]  = (_Float16)hi0.x;  a[9]  = (_Float16)hi0.y;
            a[10] = (_Float16)hi0.z;  a[11] = (_Float16)hi0.w;
            a[12] = (_Float16)hi1.x;  a[13] = (_Float16)hi1.y;
            a[14] = (_Float16)hi1.z;  a[15] = (_Float16)hi1.w;

            acc0 = __builtin_amdgcn_wmma_f32_16x16x32_f16(
                       false, a, false, Bm[0][c], (short)0, acc0, false, false);
            acc1 = __builtin_amdgcn_wmma_f32_16x16x32_f16(
                       false, a, false, Bm[1][c], (short)0, acc1, false, false);
        }

        // ---- Fused epilogue: bias + relu + final dot + sigmoid ----
        // acc element r = row (tile*16 + r + 8*ksel), column = col (+16 in acc1).
        float o[8];
        #pragma unroll
        for (int r = 0; r < 8; ++r) {
            const float h0 = fmaxf(acc0[r] + b0, 0.0f);
            const float h1 = fmaxf(acc1[r] + b1, 0.0f);
            float v = h0 * fk0 + h1 * fk1;      // this lane's share of 22-col dot
            // butterfly within each 16-lane half (wave32): sums all 22 columns
            v += __shfl_xor(v, 1);
            v += __shfl_xor(v, 2);
            v += __shfl_xor(v, 4);
            v += __shfl_xor(v, 8);
            o[r] = v;
        }

        if (col == 0) {                          // lanes 0 and 16 write 8 rows each
            float* op = out + tile * 16 + ksel * 8;
            float4 s0, s1;
            s0.x = sigmoidf(o[0] + fbias);
            s0.y = sigmoidf(o[1] + fbias);
            s0.z = sigmoidf(o[2] + fbias);
            s0.w = sigmoidf(o[3] + fbias);
            s1.x = sigmoidf(o[4] + fbias);
            s1.y = sigmoidf(o[5] + fbias);
            s1.z = sigmoidf(o[6] + fbias);
            s1.w = sigmoidf(o[7] + fbias);
            ((float4*)op)[0] = s0;
            ((float4*)op)[1] = s1;
        }
    }
}

extern "C" void kernel_launch(void* const* d_in, const int* in_sizes, int n_in,
                              void* d_out, int out_size, void* d_ws, size_t ws_size,
                              hipStream_t stream) {
    const float* x       = (const float*)d_in[0];
    const float* kernels = (const float*)d_in[1];
    const float* biases  = (const float*)d_in[2];
    const float* fk      = (const float*)d_in[3];
    const float* fb      = (const float*)d_in[4];
    float*       out     = (float*)d_out;

    const int ntiles = out_size / 16;   // 65536 row tiles of 16
    dim3 grid(2048), block(256);        // 16384 wave32 waves, 4 tiles each
    hipLaunchKernelGGL(hnn_fused_wmma, grid, block, 0, stream,
                       x, kernels, biases, fk, fb, out, ntiles);
}
